// MultiHeadAttention_30528627540393
// MI455X (gfx1250) — matrix operand order repaired and checked by reading, not compiled, not run on hardware
//
#include <hip/hip_runtime.h>
#include <math.h>

typedef __attribute__((ext_vector_type(16))) _Float16 v16h;
typedef __attribute__((ext_vector_type(8)))  float    v8f;

#define TM 128
#define TN 64
#define TK 64
#define LPAD 8
#define LSTR (TK + LPAD)   // 72 halfs per LDS row = 144B (16B aligned)

#define S_LEN 1024
#define HEADS 16
#define DEPTH 64
#define DM    1024

// ---- WMMA fragment loaders (ISA 7.12.2 layouts, wave32) -------------------

// A: 16x32 f16 (MxK). lane L: m = L&15, kbase = (L<16)?0:8;
// VGPR v holds K = kbase + (v>=4?16:0) + (v&3)*2  (+1 in high half).
// The 8 dwords form two contiguous 16B chunks at k=kb and k=kb+16:
// expressed as two uint4 loads -> guaranteed 2x ds_load_b128.
__device__ __forceinline__ v16h frag_a(const _Float16* base, int lane) {
  const int m  = lane & 15;
  const int kb = (lane >> 4) << 3;           // 0 or 8
  const _Float16* row = base + m * LSTR;
  union { uint4 u[2]; v16h v; } t;
  t.u[0] = *(const uint4*)(row + kb);        // k = kb+0,2,4,6 (+1 in hi half)
  t.u[1] = *(const uint4*)(row + kb + 16);   // k = kb+16..22
  return t.v;
}

// B: 32x16 f16 (KxN), sourced from an N-major ("Bt", N x K) LDS tile.
// lane L: n = L&15, kbase = (L<16)?0:8; K = kbase+0..7 then kbase+16..23 (matches A).
__device__ __forceinline__ v16h frag_b(const _Float16* base, int lane) {
  const int n  = lane & 15;
  const int kb = (lane >> 4) << 3;           // 0 or 8
  const _Float16* p = base + n * LSTR + kb;  // two 16B chunks at k=kb and k=kb+16, 16B aligned
  union { uint4 u[2]; v16h v; } t;
  t.u[0] = *(const uint4*)(p);
  t.u[1] = *(const uint4*)(p + 16);
  return t.v;
}

// ---- Generic f16 WMMA GEMM: C(M,N) = A(M,K) * Bt(N,K)^T, mode-based epilogue
// mode 0: proj Q/K -> f16 out at [(b*16+h)*1024+s]*64+d, +bias
// mode 1: proj V   -> f16 out at [(b*16+h)*64+d]*1024+s, +bias (transposed)
// mode 2: logits   -> f32 out at z*S*S + m*S + n; val*0.125 + mask*(-1e9) + adjoin
// mode 3: ctx      -> f16 out at (b*1024+m)*1024 + h*64 + n
// mode 4: dense    -> f32 out at m*1024 + n, +bias
__global__ __launch_bounds__(256)
void gemm_f16_wmma(const void* __restrict__ Aptr, long lda, long a_zstride, int a_f32,
                   const void* __restrict__ Bptr, long ldb, long b_zstride, int b_f32,
                   const float* __restrict__ bias,
                   void* __restrict__ Out,
                   const float* __restrict__ mask,
                   const float* __restrict__ adjoin,
                   int K, int mode)
{
  // double-buffered operand tiles: ~55 KB of the 320 KB/WGP LDS
  __shared__ _Float16 As[2][TM][LSTR];
  __shared__ _Float16 Bs[2][TN][LSTR];

  const int tid  = threadIdx.x;
  const int lane = tid & 31;
  const int wid  = tid >> 5;
  const int z    = blockIdx.z;
  const int m0   = blockIdx.y * TM;
  const int n0   = blockIdx.x * TN;

  const char* Ab = (const char*)Aptr + (size_t)z * (size_t)a_zstride * (size_t)(a_f32 ? 4 : 2);
  const char* Bb = (const char*)Bptr + (size_t)z * (size_t)b_zstride * (size_t)(b_f32 ? 4 : 2);

  const v8f vzero = {0.f, 0.f, 0.f, 0.f, 0.f, 0.f, 0.f, 0.f};
  v8f acc[2][2] = {{vzero, vzero}, {vzero, vzero}};

  const int rA = tid >> 4;         // 0..15 (row within a 16-row pass)
  const int cA = (tid & 15) << 2;  // column group of 4

  // per-wave 2x2 sub-tile of 16x16 tiles
  const int mt0 = (wid & 3) * 2;   // m-tile base (of 8)
  const int nt0 = (wid >> 2) * 2;  // n-tile base (of 4)

  // register staging for one K-tile (12 b128 loads in flight)
  float4 aF[8]; uint2 aH[8];
  float4 bF[4]; uint2 bH[4];

  auto load_tiles = [&](int k0) {
    if (a_f32) {
      const float* Ag = (const float*)Ab + (size_t)(m0 + rA) * lda + k0 + cA;
#pragma unroll
      for (int p = 0; p < 8; ++p) aF[p] = *(const float4*)(Ag + (size_t)(p * 16) * lda);
    } else {
      const _Float16* Ag = (const _Float16*)Ab + (size_t)(m0 + rA) * lda + k0 + cA;
#pragma unroll
      for (int p = 0; p < 8; ++p) aH[p] = *(const uint2*)(Ag + (size_t)(p * 16) * lda);
    }
    if (b_f32) {
      const float* Bg = (const float*)Bb + (size_t)(n0 + rA) * ldb + k0 + cA;
#pragma unroll
      for (int p = 0; p < 4; ++p) bF[p] = *(const float4*)(Bg + (size_t)(p * 16) * ldb);
    } else {
      const _Float16* Bg = (const _Float16*)Bb + (size_t)(n0 + rA) * ldb + k0 + cA;
#pragma unroll
      for (int p = 0; p < 4; ++p) bH[p] = *(const uint2*)(Bg + (size_t)(p * 16) * ldb);
    }
  };

  auto store_tiles = [&](int buf) {
    if (a_f32) {
#pragma unroll
      for (int p = 0; p < 8; ++p) {
        _Float16* d = &As[buf][p * 16 + rA][cA];
        d[0] = (_Float16)aF[p].x; d[1] = (_Float16)aF[p].y;
        d[2] = (_Float16)aF[p].z; d[3] = (_Float16)aF[p].w;
      }
    } else {
#pragma unroll
      for (int p = 0; p < 8; ++p) *(uint2*)&As[buf][p * 16 + rA][cA] = aH[p];
    }
    if (b_f32) {
#pragma unroll
      for (int p = 0; p < 4; ++p) {
        _Float16* d = &Bs[buf][p * 16 + rA][cA];
        d[0] = (_Float16)bF[p].x; d[1] = (_Float16)bF[p].y;
        d[2] = (_Float16)bF[p].z; d[3] = (_Float16)bF[p].w;
      }
    } else {
#pragma unroll
      for (int p = 0; p < 4; ++p) *(uint2*)&Bs[buf][p * 16 + rA][cA] = bH[p];
    }
  };

  const int iters = K / TK;

  // prologue: stage tile 0 into buffer 0
  load_tiles(0);
  store_tiles(0);
  __syncthreads();

  for (int it = 0; it < iters; ++it) {
    const int cur = it & 1;
    const bool has_next = (it + 1) < iters;

    if (has_next) {
      load_tiles((it + 1) * TK);   // issue next tile's global loads early
      if (it + 2 < iters) {        // pull the K+2 tile toward L2 meanwhile
        const int kp = (it + 2) * TK;
        __builtin_prefetch((const char*)Ab +
            ((size_t)(m0 + rA) * lda + kp) * (size_t)(a_f32 ? 4 : 2), 0, 0);
        __builtin_prefetch((const char*)Bb +
            ((size_t)(n0 + rA) * ldb + kp) * (size_t)(b_f32 ? 4 : 2), 0, 0);
      }
    }

    // ---- 8 WMMAs per K-tile per wave: 2x2 tiles, both fragments reused
#pragma unroll
    for (int kk = 0; kk < TK; kk += 32) {
      const v16h a0 = frag_a(&As[cur][(mt0 + 0) * 16][kk], lane);
      const v16h a1 = frag_a(&As[cur][(mt0 + 1) * 16][kk], lane);
      const v16h b0 = frag_b(&Bs[cur][(nt0 + 0) * 16][kk], lane);
      const v16h b1 = frag_b(&Bs[cur][(nt0 + 1) * 16][kk], lane);
      acc[0][0] = __builtin_amdgcn_wmma_f32_16x16x32_f16(false, a0, false, b0, (short)0, acc[0][0], false, false);
      acc[0][1] = __builtin_amdgcn_wmma_f32_16x16x32_f16(false, a0, false, b1, (short)0, acc[0][1], false, false);
      acc[1][0] = __builtin_amdgcn_wmma_f32_16x16x32_f16(false, a1, false, b0, (short)0, acc[1][0], false, false);
      acc[1][1] = __builtin_amdgcn_wmma_f32_16x16x32_f16(false, a1, false, b1, (short)0, acc[1][1], false, false);
    }

    if (has_next) store_tiles(cur ^ 1);  // write the other buffer (retired readers)
    __syncthreads();                     // one barrier per K-tile
  }

  // ---- epilogue. C/D layout: lane L -> n = L&15; VGPR r -> m = r + (L>=16?8:0)
  const int cn  = lane & 15;
  const int rm0 = (lane >> 4) << 3;
#pragma unroll
  for (int im = 0; im < 2; ++im) {
#pragma unroll
    for (int in = 0; in < 2; ++in) {
#pragma unroll
      for (int r = 0; r < 8; ++r) {
        const int m = m0 + (mt0 + im) * 16 + rm0 + r;
        const int n = n0 + (nt0 + in) * 16 + cn;
        float val = acc[im][in][r];
        if (mode == 0) {
          val += bias[n];
          const int b = m >> 10, s = m & 1023, h = n >> 6, d = n & 63;
          ((_Float16*)Out)[(((size_t)(b * HEADS + h) * S_LEN) + s) * DEPTH + d] = (_Float16)val;
        } else if (mode == 1) {
          val += bias[n];
          const int b = m >> 10, s = m & 1023, h = n >> 6, d = n & 63;
          ((_Float16*)Out)[(((size_t)(b * HEADS + h) * DEPTH) + d) * S_LEN + s] = (_Float16)val;
        } else if (mode == 2) {
          const int b = z >> 4;
          val = val * 0.125f
              + mask[(size_t)b * S_LEN + n] * (-1e9f)
              + adjoin[(size_t)b * S_LEN * S_LEN + (size_t)m * S_LEN + n];
          ((float*)Out)[(size_t)z * S_LEN * S_LEN + (size_t)m * S_LEN + n] = val;
        } else if (mode == 3) {
          const int b = z >> 4, h = z & 15;
          ((_Float16*)Out)[((size_t)(b * S_LEN + m) * DM) + h * DEPTH + n] = (_Float16)val;
        } else { // mode 4
          ((float*)Out)[(size_t)m * DM + n] = val + bias[n];
        }
      }
    }
  }
}

// ---- row softmax over S=1024, in place on the fp32 attn region ------------
__global__ __launch_bounds__(256)
void softmax_rows_kernel(float* __restrict__ attn)
{
  const size_t row = blockIdx.x;
  float* p = attn + row * S_LEN;
  const int t = threadIdx.x;
  __shared__ float red[256];

  float x0 = p[t], x1 = p[t + 256], x2 = p[t + 512], x3 = p[t + 768];
  float mx = fmaxf(fmaxf(x0, x1), fmaxf(x2, x3));
  red[t] = mx;
  __syncthreads();
  for (int s = 128; s > 0; s >>= 1) {
    if (t < s) red[t] = fmaxf(red[t], red[t + s]);
    __syncthreads();
  }
  mx = red[0];
  __syncthreads();

  x0 = __expf(x0 - mx); x1 = __expf(x1 - mx);
  x2 = __expf(x2 - mx); x3 = __expf(x3 - mx);
  red[t] = x0 + x1 + x2 + x3;
  __syncthreads();
  for (int s = 128; s > 0; s >>= 1) {
    if (t < s) red[t] += red[t + s];
    __syncthreads();
  }
  const float inv = 1.0f / red[0];
  p[t] = x0 * inv; p[t + 256] = x1 * inv;
  p[t + 512] = x2 * inv; p[t + 768] = x3 * inv;
}

// ---------------------------------------------------------------------------
extern "C" void kernel_launch(void* const* d_in, const int* in_sizes, int n_in,
                              void* d_out, int out_size, void* d_ws, size_t ws_size,
                              hipStream_t stream)
{
  const float* v_in   = (const float*)d_in[0];
  const float* k_in   = (const float*)d_in[1];
  const float* q_in   = (const float*)d_in[2];
  const float* mask   = (const float*)d_in[3];
  const float* adjoin = (const float*)d_in[4];
  const float* wq_w   = (const float*)d_in[5];
  const float* wq_b   = (const float*)d_in[6];
  const float* wk_w   = (const float*)d_in[7];
  const float* wk_b   = (const float*)d_in[8];
  const float* wv_w   = (const float*)d_in[9];
  const float* wv_b   = (const float*)d_in[10];
  const float* dn_w   = (const float*)d_in[11];
  const float* dn_b   = (const float*)d_in[12];

  float* out  = (float*)d_out;                       // [4,1024,1024]
  float* attn = out + (size_t)4 * 1024 * 1024;       // [4,16,1024,1024]

  // workspace: 4 f16 buffers of 4096*1024 each (8 MB apiece, 32 MB total)
  char* ws = (char*)d_ws;
  const size_t BUF = (size_t)4096 * 1024 * sizeof(_Float16);
  void* qh = ws;            // [B,H,S,64] f16
  void* kh = ws + BUF;      // [B,H,S,64] f16
  void* vt = ws + 2 * BUF;  // [B,H,64,S] f16 (transposed V)
  void* cc = ws + 3 * BUF;  // [B,S,1024] f16 (concat ctx)

  const dim3 blk(256);

  // Q/K/V projections: y = x @ W^T + b   (M=4096, N=1024, K=1024)
  gemm_f16_wmma<<<dim3(16, 32, 1), blk, 0, stream>>>(
      q_in, 1024, 0, 1, wq_w, 1024, 0, 1, wq_b, qh, nullptr, nullptr, 1024, 0);
  gemm_f16_wmma<<<dim3(16, 32, 1), blk, 0, stream>>>(
      k_in, 1024, 0, 1, wk_w, 1024, 0, 1, wk_b, kh, nullptr, nullptr, 1024, 0);
  gemm_f16_wmma<<<dim3(16, 32, 1), blk, 0, stream>>>(
      v_in, 1024, 0, 1, wv_w, 1024, 0, 1, wv_b, vt, nullptr, nullptr, 1024, 1);

  // logits[z] = Qh[z] @ Kh[z]^T / 8 + mask*(-1e9) + adjoin  (per z=b*16+h)
  gemm_f16_wmma<<<dim3(16, 8, 64), blk, 0, stream>>>(
      qh, 64, 65536, 0, kh, 64, 65536, 0, nullptr, attn, mask, adjoin, 64, 2);

  // softmax over the last dim, in place (also final attn output)
  softmax_rows_kernel<<<dim3(65536), blk, 0, stream>>>(attn);

  // ctx[z] = attn[z] @ V[z]  (A = fp32 attn, Bt = Vt)  M=1024, N=64, K=1024
  gemm_f16_wmma<<<dim3(1, 8, 64), blk, 0, stream>>>(
      attn, 1024, 1048576, 1, vt, 1024, 65536, 0, nullptr, cc, nullptr, nullptr, 1024, 3);

  // out = concat @ dense_w^T + dense_b   (M=4096, N=1024, K=1024)
  gemm_f16_wmma<<<dim3(16, 32, 1), blk, 0, stream>>>(
      cc, 1024, 0, 0, dn_w, 1024, 0, 1, dn_b, out, nullptr, nullptr, 1024, 4);
}
